// Encoder_73589969650063
// MI455X (gfx1250) — matrix operand order repaired and checked by reading, not compiled, not run on hardware
//
#include <hip/hip_runtime.h>
#include <hip/hip_bf16.h>
#include <stdint.h>
#include <stddef.h>

#define BB 16
#define SS 128
#define HH 512
#define CC 7
#define BSN (BB * SS) /* 2048 */

typedef __attribute__((ext_vector_type(16))) __bf16       v16bf;
typedef __attribute__((ext_vector_type(8)))  __bf16       v8bf;
typedef __attribute__((ext_vector_type(8)))  float        v8f;
typedef __attribute__((ext_vector_type(4)))  unsigned int u32x4;
typedef __attribute__((ext_vector_type(8)))  int          i32x8;
typedef __attribute__((ext_vector_type(4)))  int          i32x4;

__device__ __forceinline__ float sigmoidf_(float x) { return 1.0f / (1.0f + __expf(-x)); }

// TDM: load a 2D bf16 tile (tile_d0 elems/row, tile_d1 rows, row stride
// stride0 elems) from global to LDS byte-offset lds_off.  D# packed per
// CDNA5 ISA 8.3/8.4; groups 2/3 zero (tile_dim2..4 = 0 -> 2D).
// This toolchain exposes the 6-arg builtin: (g0 u32x4, g1 i32x8, g2 i32x4,
// g3 i32x4, extra i32x8, cpol) -- extra group zero-filled.
__device__ __forceinline__ void tdm_load_2d_bf16(unsigned lds_off,
                                                 const __bf16* gptr,
                                                 unsigned tile_d0, unsigned tile_d1,
                                                 unsigned stride0) {
    unsigned long long ga = (unsigned long long)(uintptr_t)gptr;
    u32x4 g0;
    g0[0] = 1u;                                            // count=1, user desc
    g0[1] = lds_off;                                       // lds_addr
    g0[2] = (unsigned)(ga & 0xffffffffu);                  // global_addr[31:0]
    g0[3] = (unsigned)((ga >> 32) & 0x01ffffffu) | (2u << 30); // addr[56:32] | type=2
    unsigned td0 = stride0;        // tensor_dim0 >= tile_d0 (tiles fully in-bounds)
    unsigned td1 = 1u << 20;       // tensor_dim1 huge -> OOB never triggers
    i32x8 g1;
    g1[0] = (int)(1u << 16);                               // data_size=1 (2B), mask=0
    g1[1] = (int)((td0 & 0xffffu) << 16);                  // [63:48]=tdim0 lo16
    g1[2] = (int)(((td0 >> 16) & 0xffffu) | ((td1 & 0xffffu) << 16));
    g1[3] = (int)(((td1 >> 16) & 0xffffu) | ((tile_d0 & 0xffffu) << 16));
    g1[4] = (int)(tile_d1 & 0xffffu);                      // tile_dim1, tile_dim2=0
    g1[5] = (int)stride0;                                  // tensor_dim0_stride lo32
    g1[6] = 0;
    g1[7] = 0;                                             // dim1_stride=0 (2D)
    i32x4 gz = {0, 0, 0, 0};
    i32x8 gz8 = {0, 0, 0, 0, 0, 0, 0, 0};
    __builtin_amdgcn_tensor_load_to_lds(g0, g1, gz, gz, gz8, 0);
}

// ---------------- workspace layout (bytes) ----------------
static const size_t OFF_WG   = 0;                                   // bf16 W_gate  (H*H)
static const size_t OFF_WM   = OFF_WG  + (size_t)HH * HH * 2;       // bf16 W_map   (H*H)
static const size_t OFF_NHS  = OFF_WM  + (size_t)HH * HH * 2;       // f32 nhs (B*S*H)
static const size_t OFF_HIN  = OFF_NHS + (size_t)BSN * HH * 4;      // f32 h_in accum (S*H)
static const size_t OFF_RIDX = OFF_HIN + (size_t)SS * HH * 4;       // int row_idx (S*BS)
static const size_t OFF_RANC = OFF_RIDX + (size_t)SS * BSN * 4;     // f32 row_anc (S*BS)
static const size_t OFF_RCNT = OFF_RANC + (size_t)SS * BSN * 4;     // int row_cnt (S)
static const size_t OFF_HHB  = OFF_RCNT + 1024;                     // f32 hh (3H)

// ---------------- kernels ----------------

__global__ void k_zero(float* __restrict__ p, int n) {
    int i = blockIdx.x * blockDim.x + threadIdx.x;
    if (i < n) p[i] = 0.0f;
}

__global__ void k_cvt_w(const float* __restrict__ wg, const float* __restrict__ wm,
                        __bf16* __restrict__ og, __bf16* __restrict__ om, int n) {
    int i = blockIdx.x * blockDim.x + threadIdx.x;
    if (i < n) { og[i] = (__bf16)wg[i]; om[i] = (__bf16)wm[i]; }
}

// Per-idx compaction of nonzero ancestor entries (deterministic, ballot-ordered).
__global__ void k_csr(const float* __restrict__ dep,
                      int* __restrict__ ridx, float* __restrict__ ranc,
                      int* __restrict__ rcnt) {
    int idx = blockIdx.x;
    int lane = threadIdx.x;           // 32 threads = one wave
    int count = 0;
    for (int base = 0; base < BSN; base += 32) {
        int t = base + lane;          // t = b*S + j
        int b = t >> 7, j = t & (SS - 1);
        float a = dep[(size_t)b * SS * SS + (size_t)idx * SS + j];
        bool nz = (a != 0.0f);
        unsigned mlo = (unsigned)__ballot(nz);
        int pre = __popc(mlo & ((1u << lane) - 1u));
        if (nz) {
            ridx[idx * BSN + count + pre] = t;
            ranc[idx * BSN + count + pre] = a;
        }
        count += __popc(mlo);
    }
    if (lane == 0) rcnt[idx] = count;
}

// Active-row gate/map GEMMs: A staged once per block in LDS (gather+scale+cvt),
// B weight slabs DMA'd per-wave via TDM (double buffered), bf16 WMMA compute,
// fused sigmoid(g)*m row-reduction into h_in[idx].
__global__ __launch_bounds__(128) void k_gate_map(
    const int* __restrict__ node_order, int step,
    const float* __restrict__ nhs,
    const __bf16* __restrict__ wg, const __bf16* __restrict__ wm,
    const int* __restrict__ ridx, const float* __restrict__ ranc,
    const int* __restrict__ rcnt,
    const float* __restrict__ b_gate, const float* __restrict__ b_map,
    float* __restrict__ hin) {
    __shared__ __bf16 Atile[16 * HH];               // 16 KB
    __shared__ __bf16 Bslab[4][2][2][16 * 64];      // [wave][buf][mat] 32 KB

    int idx = node_order[step];
    int cnt = rcnt[idx];
    int mtile = blockIdx.y;
    if (mtile * 16 >= cnt) return;          // uniform: EXEC stays full for WMMA

    int tid  = threadIdx.x;
    int wave = tid >> 5;
    int lane = tid & 31;
    int hi   = lane >> 4;                   // lane half (WMMA K striping)
    int ml   = lane & 15;
    int ntile = blockIdx.x * 4 + wave;      // 16-column tile
    int ncol  = ntile * 16 + ml;

    // ---- kick off first B slab DMA for this wave (wave-private buffers) ----
    const __bf16* wgbase = wg + (size_t)(ntile * 16) * HH;
    const __bf16* wmbase = wm + (size_t)(ntile * 16) * HH;
    tdm_load_2d_bf16((unsigned)(uintptr_t)&Bslab[wave][0][0][0], wgbase, 64, 16, HH);
    tdm_load_2d_bf16((unsigned)(uintptr_t)&Bslab[wave][0][1][0], wmbase, 64, 16, HH);

    // ---- stage A tile (16 x 512 bf16, anc-scaled) into LDS cooperatively ----
    {
        int row = tid >> 3;                 // 0..15
        int seg = tid & 7;                  // 64 cols each
        int grow = mtile * 16 + row;
        const float* rp = nhs;
        float sc = 0.0f;
        if (grow < cnt) {
            sc = ranc[idx * BSN + grow];
            rp = nhs + (size_t)ridx[idx * BSN + grow] * HH;
        }
        __bf16* dst = &Atile[row * HH + seg * 64];
        const float* src = rp + seg * 64;
        for (int c = 0; c < 64; c += 4) {
            float4 f = *(const float4*)(src + c);
            dst[c + 0] = (__bf16)(f.x * sc);
            dst[c + 1] = (__bf16)(f.y * sc);
            dst[c + 2] = (__bf16)(f.z * sc);
            dst[c + 3] = (__bf16)(f.w * sc);
        }
    }
    __syncthreads();

    v8f accg = {0.f,0.f,0.f,0.f,0.f,0.f,0.f,0.f};
    v8f accm = {0.f,0.f,0.f,0.f,0.f,0.f,0.f,0.f};

    // 8 K-slabs of 64; issue s+1 while computing s; TDM completes in order.
    for (int s = 0; s < 8; ++s) {
        if (s < 7) {
            int nb = (s + 1) & 1;
            tdm_load_2d_bf16((unsigned)(uintptr_t)&Bslab[wave][nb][0][0],
                             wgbase + (s + 1) * 64, 64, 16, HH);
            tdm_load_2d_bf16((unsigned)(uintptr_t)&Bslab[wave][nb][1][0],
                             wmbase + (s + 1) * 64, 64, 16, HH);
            __builtin_amdgcn_s_wait_tensorcnt(2);   // slab s landed (in-order)
        } else {
            __builtin_amdgcn_s_wait_tensorcnt(0);
        }
        const __bf16* bg = &Bslab[wave][s & 1][0][0];
        const __bf16* bm = &Bslab[wave][s & 1][1][0];
#pragma unroll
        for (int kk = 0; kk < 2; ++kk) {
            int kc = s * 64 + kk * 32;
            // A fragment: lane ml, contiguous 16 K at kc + hi*16
            v16bf a = *(const v16bf*)&Atile[ml * HH + kc + hi * 16];
            // B fragment: row ml of slab, contiguous 16 K at kk*32 + hi*16
            v16bf fbg = *(const v16bf*)(bg + ml * 64 + kk * 32 + hi * 16);
            v16bf fbm = *(const v16bf*)(bm + ml * 64 + kk * 32 + hi * 16);
            accg = __builtin_amdgcn_wmma_f32_16x16x32_bf16(false, a, false, fbg,
                                                           (short)0, accg, false, false);
            accm = __builtin_amdgcn_wmma_f32_16x16x32_bf16(false, a, false, fbm,
                                                           (short)0, accm, false, false);
        }
    }

    // Epilogue: h_in[ncol] += sum over valid rows of sigmoid(g+bg)*(m+bm)
    float bgv = b_gate[ncol];
    float bmv = b_map[ncol];
    float part = 0.0f;
#pragma unroll
    for (int v = 0; v < 8; ++v) {
        int rowg = mtile * 16 + v + 8 * hi;     // C/D layout: VGPR v -> row v+8*hi
        if (rowg < cnt) {
            float g = sigmoidf_(accg[v] + bgv);
            float m = accm[v] + bmv;
            part += g * m;
        }
    }
    part += __shfl_xor(part, 16, 32);
    if (hi == 0) atomicAdd(&hin[(size_t)idx * HH + ncol], part);
}

// hh = (h_in_total) @ W_hh.T + b_hh  (3H outputs, each a 512-dot)
__global__ __launch_bounds__(256) void k_hh(
    const int* __restrict__ node_order, int step,
    const float* __restrict__ hin, const int* __restrict__ rcnt,
    const float* __restrict__ b_gate, const float* __restrict__ b_map,
    const float* __restrict__ W_hh, const float* __restrict__ b_hh,
    float* __restrict__ hhout) {
    __shared__ float hs[HH];
    int idx = node_order[step];
    float kz = (float)(BSN - rcnt[idx]);      // zero-row closed-form count
    for (int t = threadIdx.x; t < HH; t += blockDim.x)
        hs[t] = hin[(size_t)idx * HH + t] + kz * sigmoidf_(b_gate[t]) * b_map[t];
    __syncthreads();
    int o = blockIdx.x * blockDim.x + threadIdx.x;
    if (o < 3 * HH) {
        const float* wr = W_hh + (size_t)o * HH;
        float acc = b_hh[o];
        for (int k = 0; k < HH; k += 4)
            acc += hs[k] * wr[k] + hs[k+1] * wr[k+1] +
                   hs[k+2] * wr[k+2] + hs[k+3] * wr[k+3];
        hhout[o] = acc;
    }
}

// GRU elementwise update + scatter hv into nhs[:, idx]
__global__ __launch_bounds__(256) void k_upd(
    const int* __restrict__ node_order, int step,
    const float* __restrict__ hin, const int* __restrict__ rcnt,
    const float* __restrict__ b_gate, const float* __restrict__ b_map,
    const float* __restrict__ enc,
    const float* __restrict__ W_ih, const float* __restrict__ b_ih,
    const float* __restrict__ hhv, float* __restrict__ nhs) {
    int t = blockIdx.x * blockDim.x + threadIdx.x;
    if (t >= BB * HH) return;
    int b = t / HH, h = t % HH;
    int idx = node_order[step];
    const float* xr = enc + ((size_t)b * SS + idx) * CC;
    float x[CC];
#pragma unroll
    for (int c = 0; c < CC; ++c) x[c] = xr[c];
    float xi_r = b_ih[h], xi_z = b_ih[HH + h], xi_n = b_ih[2 * HH + h];
    const float* w0 = W_ih + (size_t)h * CC;
    const float* w1 = W_ih + (size_t)(HH + h) * CC;
    const float* w2 = W_ih + (size_t)(2 * HH + h) * CC;
#pragma unroll
    for (int c = 0; c < CC; ++c) {
        xi_r += x[c] * w0[c]; xi_z += x[c] * w1[c]; xi_n += x[c] * w2[c];
    }
    float kz = (float)(BSN - rcnt[idx]);
    float hint = hin[(size_t)idx * HH + h] + kz * sigmoidf_(b_gate[h]) * b_map[h];
    float r = sigmoidf_(xi_r + hhv[h]);
    float z = sigmoidf_(xi_z + hhv[HH + h]);
    float n = tanhf(xi_n + r * hhv[2 * HH + h]);
    float hv = (1.0f - z) * n + z * hint;
    nhs[((size_t)b * SS + idx) * HH + h] = hv;
}

// out = [hv | mu | logvar]
__global__ __launch_bounds__(256) void k_final(
    const int* __restrict__ node_order, const float* __restrict__ nhs,
    const float* __restrict__ W_mu, const float* __restrict__ b_mu,
    const float* __restrict__ W_lv, const float* __restrict__ b_lv,
    float* __restrict__ out) {
    int t = blockIdx.x * blockDim.x + threadIdx.x;
    if (t >= 3 * BB * HH) return;
    int region = t / (BB * HH);
    int r = t % (BB * HH);
    int b = r / HH, h = r % HH;
    int idx = node_order[SS - 1];
    const float* hv = nhs + ((size_t)b * SS + idx) * HH;
    if (region == 0) { out[t] = hv[h]; return; }
    const float* W  = (region == 1) ? W_mu : W_lv;
    const float* bv = (region == 1) ? b_mu : b_lv;
    float acc = bv[h];
    const float* wr = W + (size_t)h * HH;
    for (int k = 0; k < HH; ++k) acc += hv[k] * wr[k];
    out[t] = acc;
}

// ---------------- host ----------------
extern "C" void kernel_launch(void* const* d_in, const int* in_sizes, int n_in,
                              void* d_out, int out_size, void* d_ws, size_t ws_size,
                              hipStream_t stream) {
    (void)in_sizes; (void)n_in; (void)out_size; (void)ws_size;
    const float* dep    = (const float*)d_in[0];
    const float* enc    = (const float*)d_in[1];
    const int*   order  = (const int*)d_in[2];
    const float* W_gate = (const float*)d_in[3];
    const float* b_gate = (const float*)d_in[4];
    const float* W_map  = (const float*)d_in[5];
    const float* b_map  = (const float*)d_in[6];
    const float* W_ih   = (const float*)d_in[7];
    const float* b_ih   = (const float*)d_in[8];
    const float* W_hh   = (const float*)d_in[9];
    const float* b_hh   = (const float*)d_in[10];
    const float* W_mu   = (const float*)d_in[11];
    const float* b_mu   = (const float*)d_in[12];
    const float* W_lv   = (const float*)d_in[13];
    const float* b_lv   = (const float*)d_in[14];

    char* ws = (char*)d_ws;
    __bf16* wg_bf = (__bf16*)(ws + OFF_WG);
    __bf16* wm_bf = (__bf16*)(ws + OFF_WM);
    float*  nhs   = (float*)(ws + OFF_NHS);
    float*  hin   = (float*)(ws + OFF_HIN);
    int*    ridx  = (int*)(ws + OFF_RIDX);
    float*  ranc  = (float*)(ws + OFF_RANC);
    int*    rcnt  = (int*)(ws + OFF_RCNT);
    float*  hhv   = (float*)(ws + OFF_HHB);

    // nhs and hin are contiguous: zero both (must re-init every call).
    int zn = BSN * HH + SS * HH;
    hipLaunchKernelGGL(k_zero, dim3((zn + 255) / 256), dim3(256), 0, stream, nhs, zn);
    hipLaunchKernelGGL(k_cvt_w, dim3((HH * HH + 255) / 256), dim3(256), 0, stream,
                       W_gate, W_map, wg_bf, wm_bf, HH * HH);
    hipLaunchKernelGGL(k_csr, dim3(SS), dim3(32), 0, stream, dep, ridx, ranc, rcnt);

    for (int step = 0; step < SS; ++step) {
        hipLaunchKernelGGL(k_gate_map, dim3(HH / 64, BSN / 16), dim3(128), 0, stream,
                           order, step, nhs, wg_bf, wm_bf, ridx, ranc, rcnt,
                           b_gate, b_map, hin);
        hipLaunchKernelGGL(k_hh, dim3(6), dim3(256), 0, stream,
                           order, step, hin, rcnt, b_gate, b_map, W_hh, b_hh, hhv);
        hipLaunchKernelGGL(k_upd, dim3(BB * HH / 256), dim3(256), 0, stream,
                           order, step, hin, rcnt, b_gate, b_map, enc, W_ih, b_ih,
                           hhv, nhs);
    }
    hipLaunchKernelGGL(k_final, dim3(3 * BB * HH / 256), dim3(256), 0, stream,
                       order, nhs, W_mu, b_mu, W_lv, b_lv, (float*)d_out);
}
